// E3RelaxH2_58007828300073
// MI455X (gfx1250) — matrix operand order repaired and checked by reading, not compiled, not run on hardware
//
#include <hip/hip_runtime.h>
#include <hip/hip_bf16.h>

typedef __attribute__((ext_vector_type(16))) _Float16 v16h;
typedef __attribute__((ext_vector_type(8)))  _Float16 v8h;
typedef __attribute__((ext_vector_type(8)))  float    v8f;

constexpr int Hc = 128, Rc = 128, Lc = 3;
constexpr int Nc = 20000, Ec = 256000, Gc = 200;
constexpr float APGc = 100.0f;
constexpr float CUT = 6.0f;
constexpr float INV_SQRT2 = 0.70710678118654752440f;

// ---------------------------------------------------------------------------
// DimeNet-style polynomial-envelope Gaussian RBF (p=5), ds = d / cutoff
// ---------------------------------------------------------------------------
__device__ __forceinline__ float rbf_eval(float ds, int r) {
  float env = 0.0f;
  if (ds < 1.0f) {
    float d2 = ds * ds;
    float d5 = d2 * d2 * ds;
    float d6 = d5 * ds;
    float d7 = d6 * ds;
    env = 1.0f - 21.0f * d5 + 35.0f * d6 - 15.0f * d7;
  }
  float off = (float)r * (1.0f / 127.0f);
  float t = (ds - off) * 128.0f;
  return env * __expf(-0.5f * t * t);
}

__device__ __forceinline__ float silu_f(float v) {
  return v / (1.0f + __expf(-v));
}

// ---------------------------------------------------------------------------
// Panelized WMMA GEMM:  C[M,N] = act(A[M,K](f16) * B[K,N](f16) + bias[N])
// Block: 256 threads = 8 waves; tile 32 rows x 128 cols.
// Whole 128-wide K panel staged in LDS once; A panel staged with
// global_load_async_to_lds_b128 (ASYNCcnt); then 4 K-steps x 2 WMMAs per wave
// with no intra-panel barriers. K must be a multiple of 128 (128/256 here),
// N a multiple of 128 (128/384 here).
// ---------------------------------------------------------------------------
template <bool SILU>
__global__ __launch_bounds__(256)
void gemm_wmma(const _Float16* __restrict__ A, const _Float16* __restrict__ B,
               const float* __restrict__ bias, float* __restrict__ Cf,
               _Float16* __restrict__ Ch, int M, int N, int K) {
  constexpr int KP = 128;    // K panel width
  constexpr int LDA = 136;   // padded LDS strides (byte stride 272 = 17*16)
  const int tid  = threadIdx.x;
  const int wave = tid >> 5;
  const int lane = tid & 31;
  const int half = lane >> 4;
  const int l16  = lane & 15;
  const int row0 = blockIdx.x * 32;
  const int col0 = blockIdx.y * 128;

  __shared__ _Float16 As[32 * LDA];    // 32 rows x 128 k
  __shared__ _Float16 Bs[128 * LDA];   // transposed: Bs[n][k]

  v8f acc0 = {}, acc1 = {};

  for (int kp = 0; kp < K; kp += KP) {
    // ---- stage A panel (32 x 128 halfs) via async global->LDS, 32B/thread --
    {
      int r = tid >> 3;            // 0..31
      int c = (tid & 7) * 16;      // 0..112 (halfs)
      int gr = row0 + r;
      int gra = (gr < M) ? gr : (M - 1);   // clamp: junk rows never stored
      const _Float16* gp = A + (size_t)gra * K + kp + c;
      unsigned lds0 = (unsigned)(size_t)(const void*)&As[r * LDA + c];
      unsigned long long ga = (unsigned long long)(size_t)gp;
      asm volatile("global_load_async_to_lds_b128 %0, %1, off"
                   :: "v"(lds0), "v"(ga) : "memory");
      asm volatile("global_load_async_to_lds_b128 %0, %1, off"
                   :: "v"(lds0 + 16u), "v"(ga + 16ull) : "memory");
    }
    // ---- stage B panel (128k x 128n) transposed with b32 repack ----
#pragma unroll
    for (int it = 0; it < 4; ++it) {
      int tt = tid + it * 256;        // 0..1023
      int kk = (tt & 15) * 8;         // 0..120
      int nb = (tt >> 4) * 2;         // 0..126
      const _Float16* gp = B + (size_t)(kp + kk) * N + col0 + nb;
      unsigned w[8];
#pragma unroll
      for (int m = 0; m < 8; ++m)
        w[m] = *(const unsigned*)(gp + (size_t)m * N);
#pragma unroll
      for (int m = 0; m < 4; ++m) {
        unsigned e0 = w[2 * m], e1 = w[2 * m + 1];
        *(unsigned*)&Bs[nb * LDA + kk + 2 * m] =
            (e0 & 0xffffu) | (e1 << 16);
        *(unsigned*)&Bs[(nb + 1) * LDA + kk + 2 * m] =
            (e0 >> 16) | (e1 & 0xffff0000u);
      }
    }
    asm volatile("s_wait_asynccnt 0x0" ::: "memory");
    __syncthreads();

    // ---- 4 K-steps, 2 WMMAs each, barrier-free ----
#pragma unroll
    for (int ks = 0; ks < KP; ks += 32) {
      const _Float16* a0p = &As[l16 * LDA + ks + half * 8];
      const _Float16* a1p = a0p + 16 * LDA;
      const _Float16* bp  = &Bs[(wave * 16 + l16) * LDA + ks + half * 8];
      v8h x0 = *(const v8h*)a0p;
      v8h x1 = *(const v8h*)(a0p + 16);
      v16h af0 = __builtin_shufflevector(x0, x1, 0, 1, 2, 3, 4, 5, 6, 7,
                                         8, 9, 10, 11, 12, 13, 14, 15);
      v8h y0 = *(const v8h*)a1p;
      v8h y1 = *(const v8h*)(a1p + 16);
      v16h af1 = __builtin_shufflevector(y0, y1, 0, 1, 2, 3, 4, 5, 6, 7,
                                         8, 9, 10, 11, 12, 13, 14, 15);
      v8h b0 = *(const v8h*)bp;
      v8h b1 = *(const v8h*)(bp + 16);
      v16h bf = __builtin_shufflevector(b0, b1, 0, 1, 2, 3, 4, 5, 6, 7,
                                        8, 9, 10, 11, 12, 13, 14, 15);
      acc0 = __builtin_amdgcn_wmma_f32_16x16x32_f16(false, af0, false, bf,
                                                    (short)0, acc0, false, false);
      acc1 = __builtin_amdgcn_wmma_f32_16x16x32_f16(false, af1, false, bf,
                                                    (short)0, acc1, false, false);
    }
    __syncthreads();
  }

  // D layout: acc[r] -> row = base + r + 8*half, col = col0 + wave*16 + l16
  const int colg = col0 + wave * 16 + l16;
  float bval = bias ? bias[colg] : 0.0f;
#pragma unroll
  for (int r = 0; r < 8; ++r) {
    int rowg0 = row0 + r + half * 8;
    int rowg1 = rowg0 + 16;
    if (rowg0 < M) {
      float v = acc0[r] + bval;
      if (SILU) v = silu_f(v);
      size_t o = (size_t)rowg0 * N + colg;
      if (Cf) Cf[o] = v;
      if (Ch) Ch[o] = (_Float16)v;
    }
    if (rowg1 < M) {
      float v = acc1[r] + bval;
      if (SILU) v = silu_f(v);
      size_t o = (size_t)rowg1 * N + colg;
      if (Cf) Cf[o] = v;
      if (Ch) Ch[o] = (_Float16)v;
    }
  }
}

// ---------------------------------------------------------------------------
// Small utility kernels
// ---------------------------------------------------------------------------
__global__ void k_zero(float* p, long long n) {
  long long i = (long long)blockIdx.x * blockDim.x + threadIdx.x;
  if (i < n) p[i] = 0.0f;
}
__global__ void k_copy(float* d, const float* s, long long n) {
  long long i = (long long)blockIdx.x * blockDim.x + threadIdx.x;
  if (i < n) d[i] = s[i];
}
__global__ void k_scale_f16(const float* s, _Float16* d, float sc, long long n) {
  long long i = (long long)blockIdx.x * blockDim.x + threadIdx.x;
  if (i < n) d[i] = (_Float16)(s[i] * sc);
}
__global__ void k_init_atom_s(const int* z, const float* emb, float* as, long long n) {
  long long i = (long long)blockIdx.x * blockDim.x + threadIdx.x;
  if (i < n) {
    int a = (int)(i >> 7), h = (int)(i & 127);
    as[i] = emb[z[a] * Hc + h];
  }
}
__global__ void k_init_lat_s(const float* lat_emb, float* lat_s, long long n) {
  long long i = (long long)blockIdx.x * blockDim.x + threadIdx.x;
  if (i < n) {
    int v = (int)((i >> 7) % 3), h = (int)(i & 127);
    lat_s[i] = lat_emb[v * Hc + h];
  }
}

// ---------------------------------------------------------------------------
// Edge geometry + RBF: one block (128 threads) per edge
// ---------------------------------------------------------------------------
__global__ void k_edge_geom(const float* __restrict__ pos, const float* __restrict__ cell,
                            const float* __restrict__ offs, const int* __restrict__ eidx,
                            const int* __restrict__ batch, float* __restrict__ udiff,
                            _Float16* __restrict__ efeat) {
  int e = blockIdx.x, h = threadIdx.x;
  __shared__ float s_ds;
  if (h == 0) {
    int j = eidx[e], i = eidx[Ec + e];
    int g = batch[i];
    float o0 = offs[e * 3 + 0], o1 = offs[e * 3 + 1], o2 = offs[e * 3 + 2];
    float pd[3];
#pragma unroll
    for (int d = 0; d < 3; ++d) {
      float sh = o0 * cell[g * 9 + d] + o1 * cell[g * 9 + 3 + d] + o2 * cell[g * 9 + 6 + d];
      pd[d] = pos[j * 3 + d] + sh - pos[i * 3 + d];
    }
    float dist = sqrtf(pd[0] * pd[0] + pd[1] * pd[1] + pd[2] * pd[2]);
    float inv = -1.0f / (dist + 1e-9f);
#pragma unroll
    for (int d = 0; d < 3; ++d) udiff[e * 3 + d] = pd[d] * inv;
    s_ds = dist * (1.0f / CUT);
  }
  __syncthreads();
  efeat[(size_t)e * Rc + h] = (_Float16)rbf_eval(s_ds, h);
}

// atom -> per-graph position sum (mean applied at use site)
__global__ void k_pos_accum(const float* pos, const int* batch, float* psum, long long n) {
  long long a = (long long)blockIdx.x * blockDim.x + threadIdx.x;
  if (a < n) {
    int g = batch[a];
#pragma unroll
    for (int d = 0; d < 3; ++d) atomicAdd(&psum[g * 3 + d], pos[a * 3 + d]);
  }
}

// ---------------------------------------------------------------------------
// Lattice per-atom features: one block per (atom, axis)
// ---------------------------------------------------------------------------
__global__ void k_lat_atom(const float* __restrict__ pos, const float* __restrict__ psum,
                           const float* __restrict__ cell, const int* __restrict__ batch,
                           float* __restrict__ lat_udiff, _Float16* __restrict__ latrbf) {
  int av = blockIdx.x;
  int a = av / 3, v = av - a * 3;
  int h = threadIdx.x;
  __shared__ float s_para, s_perp;
  if (h == 0) {
    int g = batch[a];
    float pc[3], row[3];
#pragma unroll
    for (int d = 0; d < 3; ++d) {
      pc[d] = pos[a * 3 + d] - psum[g * 3 + d] * (1.0f / APGc);
      row[d] = cell[g * 9 + v * 3 + d];
    }
    float cn = sqrtf(row[0] * row[0] + row[1] * row[1] + row[2] * row[2]);
    float inv = 1.0f / (cn + 1e-9f);
    float dpara = (pc[0] * row[0] + pc[1] * row[1] + pc[2] * row[2]) * inv;
    float cx = pc[1] * row[2] - pc[2] * row[1];
    float cy = pc[2] * row[0] - pc[0] * row[2];
    float cz = pc[0] * row[1] - pc[1] * row[0];
    float dperp = sqrtf(cx * cx + cy * cy + cz * cz) * inv;
    float ld[3];
#pragma unroll
    for (int d = 0; d < 3; ++d) ld[d] = row[d] - pc[d];
    float ldist = sqrtf(ld[0] * ld[0] + ld[1] * ld[1] + ld[2] * ld[2]);
    float li = -1.0f / (ldist + 1e-9f);
#pragma unroll
    for (int d = 0; d < 3; ++d) lat_udiff[a * 9 + v * 3 + d] = ld[d] * li;
    s_para = dpara * (1.0f / CUT);
    s_perp = dperp * (1.0f / CUT);
  }
  __syncthreads();
  latrbf[(size_t)av * 256 + h]       = (_Float16)rbf_eval(s_para, h);
  latrbf[(size_t)av * 256 + 128 + h] = (_Float16)rbf_eval(s_perp, h);
}

// ---------------------------------------------------------------------------
// Lattice block apply: one block (128 threads = h) per atom
// ---------------------------------------------------------------------------
__global__ void k_lb_apply(const float* __restrict__ filt, const float* __restrict__ phi_lb,
                           const float* __restrict__ lat_v, const float* __restrict__ lat_udiff,
                           const int* __restrict__ batch,
                           float* __restrict__ atom_s, float* __restrict__ atom_v) {
  int a = blockIdx.x, h = threadIdx.x;
  int g = batch[a];
  float acc_s = 0.0f, accv0 = 0.0f, accv1 = 0.0f, accv2 = 0.0f;
#pragma unroll
  for (int v = 0; v < 3; ++v) {
    size_t fr = (size_t)(a * 3 + v) * 384;
    size_t pr = (size_t)(g * 3 + v) * 384;
    float s1 = filt[fr + h]       * phi_lb[pr + h];
    float s2 = filt[fr + 128 + h] * phi_lb[pr + 128 + h];
    float s3 = filt[fr + 256 + h] * phi_lb[pr + 256 + h];
    acc_s += s1;
    accv0 += lat_v[((size_t)(g * 3 + v) * 3 + 0) * Hc + h] * s2 + lat_udiff[a * 9 + v * 3 + 0] * s3;
    accv1 += lat_v[((size_t)(g * 3 + v) * 3 + 1) * Hc + h] * s2 + lat_udiff[a * 9 + v * 3 + 1] * s3;
    accv2 += lat_v[((size_t)(g * 3 + v) * 3 + 2) * Hc + h] * s2 + lat_udiff[a * 9 + v * 3 + 2] * s3;
  }
  atom_s[(size_t)a * Hc + h] += acc_s;
  atom_v[((size_t)a * 3 + 0) * Hc + h] += accv0;
  atom_v[((size_t)a * 3 + 1) * Hc + h] += accv1;
  atom_v[((size_t)a * 3 + 2) * Hc + h] += accv2;
}

// ---------------------------------------------------------------------------
// Message passing scatter: one block (128 threads = h) per edge
// ---------------------------------------------------------------------------
__global__ void k_message(const float* __restrict__ phi, const _Float16* __restrict__ Wf,
                          const float* __restrict__ atom_v, const float* __restrict__ udiff,
                          const int* __restrict__ eidx,
                          float* __restrict__ d_s, float* __restrict__ d_v) {
  int e = blockIdx.x, h = threadIdx.x;
  int j = eidx[e], i = eidx[Ec + e];
  size_t wr = (size_t)e * 384;
  size_t pr = (size_t)j * 384;
  __builtin_prefetch(Wf + wr + 384, 0, 1);   // speculative: next edge's filter row
  float ma = phi[pr + h]       * (float)Wf[wr + h];
  float mb = phi[pr + 128 + h] * (float)Wf[wr + 128 + h];
  float mc = phi[pr + 256 + h] * (float)Wf[wr + 256 + h];
  atomicAdd(&d_s[(size_t)i * Hc + h], ma);
#pragma unroll
  for (int d = 0; d < 3; ++d) {
    float m = atom_v[((size_t)j * 3 + d) * Hc + h] * mb + udiff[e * 3 + d] * mc;
    atomicAdd(&d_v[((size_t)i * 3 + d) * Hc + h], m);
  }
}

__global__ void k_mp_apply_s(float* as, const float* ds, long long n) {
  long long i = (long long)blockIdx.x * blockDim.x + threadIdx.x;
  if (i < n) as[i] = (as[i] + ds[i]) * INV_SQRT2;
}
__global__ void k_add(float* a, const float* b, long long n) {
  long long i = (long long)blockIdx.x * blockDim.x + threadIdx.x;
  if (i < n) a[i] += b[i];
}

// ---------------------------------------------------------------------------
// Self-interaction update helpers
// ---------------------------------------------------------------------------
__global__ void k_vnorm_cat(const float* __restrict__ as, const float* __restrict__ Vv,
                            _Float16* __restrict__ cat, long long n) {
  long long i = (long long)blockIdx.x * blockDim.x + threadIdx.x;
  if (i < n) {
    long long a = i >> 7;
    int h = (int)(i & 127);
    float v0 = Vv[(a * 3 + 0) * Hc + h];
    float v1 = Vv[(a * 3 + 1) * Hc + h];
    float v2 = Vv[(a * 3 + 2) * Hc + h];
    float vn = sqrtf(v0 * v0 + v1 * v1 + v2 * v2 + 1e-9f);
    cat[a * 256 + h]       = (_Float16)as[i];
    cat[a * 256 + 128 + h] = (_Float16)vn;
  }
}

__global__ void k_update_apply(float* __restrict__ as, float* __restrict__ av,
                               const float* __restrict__ Uv, const float* __restrict__ Vv,
                               const float* __restrict__ m, long long n) {
  long long i = (long long)blockIdx.x * blockDim.x + threadIdx.x;
  if (i < n) {
    long long a = i >> 7;
    int h = (int)(i & 127);
    float dot = 0.0f, u[3];
#pragma unroll
    for (int d = 0; d < 3; ++d) {
      u[d] = Uv[(a * 3 + d) * Hc + h];
      dot += u[d] * Vv[(a * 3 + d) * Hc + h];
    }
    float ass = m[a * 384 + h];
    float asv = m[a * 384 + 128 + h];
    float avv = m[a * 384 + 256 + h];
    as[i] = (as[i] + ass + asv * dot) * INV_SQRT2;
#pragma unroll
    for (int d = 0; d < 3; ++d) av[(a * 3 + d) * Hc + h] += avv * u[d];
  }
}

// pos[a][d] += dot(atom_v[a][d][:], su_w)
__global__ void k_structure(float* __restrict__ pos, const float* __restrict__ av,
                            const float* __restrict__ suw, long long n) {
  long long i = (long long)blockIdx.x * blockDim.x + threadIdx.x;
  if (i < n) {
    const float* vr = av + i * Hc;
    float acc = 0.0f;
    for (int h = 0; h < Hc; ++h) acc += vr[h] * suw[h];
    pos[i] += acc;
  }
}

// ---------------------------------------------------------------------------
// Graph pooling & lattice decode
// ---------------------------------------------------------------------------
__global__ void k_pool(const float* __restrict__ as, const float* __restrict__ av,
                       const int* __restrict__ batch, float* __restrict__ spool,
                       float* __restrict__ vpool, long long n) {
  long long i = (long long)blockIdx.x * blockDim.x + threadIdx.x;
  if (i < n) {
    long long a = i >> 7;
    int h = (int)(i & 127);
    int g = batch[a];
    atomicAdd(&spool[(size_t)g * Hc + h], as[i]);
#pragma unroll
    for (int d = 0; d < 3; ++d)
      atomicAdd(&vpool[((size_t)g * 3 + d) * Hc + h], av[(a * 3 + d) * Hc + h]);
  }
}

__global__ void k_lat_apply(float* __restrict__ lat_s, float* __restrict__ lat_v,
                            float* __restrict__ cell, const float* __restrict__ t_ls,
                            const float* __restrict__ vproj, const float* __restrict__ ldec) {
  int g = blockIdx.x, h = threadIdx.x;
  float ts = t_ls[(size_t)g * Hc + h];
#pragma unroll
  for (int v = 0; v < 3; ++v) lat_s[((size_t)g * 3 + v) * Hc + h] += ts;
#pragma unroll
  for (int a = 0; a < 3; ++a)
#pragma unroll
    for (int d = 0; d < 3; ++d)
      lat_v[(((size_t)g * 3 + a) * 3 + d) * Hc + h] += vproj[((size_t)g * 3 + d) * Hc + h];
  __syncthreads();
  __shared__ float red[128];
  float w = ldec[h];
  for (int ad = 0; ad < 9; ++ad) {
    int a = ad / 3, d = ad % 3;
    red[h] = lat_v[(((size_t)g * 3 + a) * 3 + d) * Hc + h] * w;
    __syncthreads();
    for (int s = 64; s > 0; s >>= 1) {
      if (h < s) red[h] += red[h + s];
      __syncthreads();
    }
    if (h == 0) cell[g * 9 + ad] += red[0];
    __syncthreads();
  }
}

// ===========================================================================
// Host-side orchestration
// ===========================================================================
extern "C" void kernel_launch(void* const* d_in, const int* in_sizes, int n_in,
                              void* d_out, int out_size, void* d_ws, size_t ws_size,
                              hipStream_t stream) {
  (void)in_sizes; (void)n_in; (void)out_size; (void)ws_size;
  const int*   z        = (const int*)  d_in[0];
  const float* pos0     = (const float*)d_in[1];
  const float* cell0    = (const float*)d_in[2];
  const float* offs     = (const float*)d_in[3];
  const int*   eidx     = (const int*)  d_in[4];
  const int*   batch    = (const int*)  d_in[5];
  const float* atom_emb = (const float*)d_in[6];
  const float* lat_emb  = (const float*)d_in[7];
  const float* fm_w1 = (const float*)d_in[8];   const float* fm_b1 = (const float*)d_in[9];
  const float* fm_w2 = (const float*)d_in[10];  const float* fm_b2 = (const float*)d_in[11];
  const float* mp_wp1 = (const float*)d_in[12]; const float* mp_bp1 = (const float*)d_in[13];
  const float* mp_wp2 = (const float*)d_in[14]; const float* mp_bp2 = (const float*)d_in[15];
  const float* mp_wf = (const float*)d_in[16];  const float* mp_bf = (const float*)d_in[17];
  const float* up_u = (const float*)d_in[18];   const float* up_v = (const float*)d_in[19];
  const float* up_w1 = (const float*)d_in[20];  const float* up_b1 = (const float*)d_in[21];
  const float* up_w2 = (const float*)d_in[22];  const float* up_b2 = (const float*)d_in[23];
  const float* su_w = (const float*)d_in[24];
  const float* lb_wf = (const float*)d_in[25];  const float* lb_bf = (const float*)d_in[26];
  const float* lb_wp1 = (const float*)d_in[27]; const float* lb_bp1 = (const float*)d_in[28];
  const float* lb_wp2 = (const float*)d_in[29]; const float* lb_bp2 = (const float*)d_in[30];
  const float* lg_ws = (const float*)d_in[31];  const float* lg_bs = (const float*)d_in[32];
  const float* lg_wv = (const float*)d_in[33];
  const float* ldec_w = (const float*)d_in[34];
  float* out = (float*)d_out;

  // workspace bump allocator (256B aligned)
  size_t off = 0;
  auto alloc = [&](size_t bytes) -> void* {
    void* p = (char*)d_ws + off;
    off += (bytes + 255) & ~(size_t)255;
    return p;
  };
  float*     pos_cur  = (float*)alloc(sizeof(float) * Nc * 3);
  float*     cell_cur = (float*)alloc(sizeof(float) * Gc * 9);
  float*     atom_s   = (float*)alloc(sizeof(float) * Nc * Hc);
  float*     atom_v   = (float*)alloc(sizeof(float) * Nc * 3 * Hc);
  float*     lat_s    = (float*)alloc(sizeof(float) * Gc * 3 * Hc);
  float*     lat_v    = (float*)alloc(sizeof(float) * Gc * 9 * Hc);
  float*     pos_sum  = (float*)alloc(sizeof(float) * Gc * 3);
  float*     udiff    = (float*)alloc(sizeof(float) * Ec * 3);
  float*     lat_ud   = (float*)alloc(sizeof(float) * Nc * 9);
  _Float16*  efeat_h  = (_Float16*)alloc(sizeof(_Float16) * (size_t)Ec * Rc);
  _Float16*  latrbf_h = (_Float16*)alloc(sizeof(_Float16) * (size_t)Nc * 3 * 256);
  _Float16*  lat_h1_h = (_Float16*)alloc(sizeof(_Float16) * (size_t)Nc * 3 * Rc);
  _Float16*  lat_ft_h = (_Float16*)alloc(sizeof(_Float16) * (size_t)Nc * 3 * Rc);
  float*     filt     = (float*)alloc(sizeof(float) * (size_t)Nc * 3 * 384);
  _Float16*  lats_h   = (_Float16*)alloc(sizeof(_Float16) * Gc * 3 * Hc);
  _Float16*  lbp_t_h  = (_Float16*)alloc(sizeof(_Float16) * Gc * 3 * Hc);
  float*     phi_lb   = (float*)alloc(sizeof(float) * Gc * 3 * 384);
  _Float16*  atoms_h  = (_Float16*)alloc(sizeof(_Float16) * Nc * Hc);
  _Float16*  mp_t_h   = (_Float16*)alloc(sizeof(_Float16) * Nc * Hc);
  float*     phi_mp   = (float*)alloc(sizeof(float) * (size_t)Nc * 384);
  _Float16*  Wf_h     = (_Float16*)alloc(sizeof(_Float16) * (size_t)Ec * 384);
  float*     d_s      = (float*)alloc(sizeof(float) * Nc * Hc);
  float*     d_v      = (float*)alloc(sizeof(float) * Nc * 3 * Hc);
  _Float16*  atomv_h  = (_Float16*)alloc(sizeof(_Float16) * Nc * 3 * Hc);
  float*     Uv       = (float*)alloc(sizeof(float) * Nc * 3 * Hc);
  float*     Vv       = (float*)alloc(sizeof(float) * Nc * 3 * Hc);
  _Float16*  cat_h    = (_Float16*)alloc(sizeof(_Float16) * Nc * 256);
  _Float16*  up_t_h   = (_Float16*)alloc(sizeof(_Float16) * Nc * Hc);
  float*     m_up     = (float*)alloc(sizeof(float) * (size_t)Nc * 384);
  float*     s_pool   = (float*)alloc(sizeof(float) * Gc * Hc);
  float*     v_pool   = (float*)alloc(sizeof(float) * Gc * 3 * Hc);
  _Float16*  spool_h  = (_Float16*)alloc(sizeof(_Float16) * Gc * Hc);
  _Float16*  vpool_h  = (_Float16*)alloc(sizeof(_Float16) * Gc * 3 * Hc);
  float*     t_ls     = (float*)alloc(sizeof(float) * Gc * Hc);
  float*     vproj    = (float*)alloc(sizeof(float) * Gc * 3 * Hc);
  // f16 weight copies
  _Float16* fm_w1_h  = (_Float16*)alloc(sizeof(_Float16) * Lc * 256 * 128);
  _Float16* fm_w2_h  = (_Float16*)alloc(sizeof(_Float16) * Lc * 128 * 128);
  _Float16* lb_wf_h  = (_Float16*)alloc(sizeof(_Float16) * Lc * 128 * 384);
  _Float16* lb_wp1_h = (_Float16*)alloc(sizeof(_Float16) * Lc * 128 * 128);
  _Float16* lb_wp2_h = (_Float16*)alloc(sizeof(_Float16) * Lc * 128 * 384);
  _Float16* mp_wp1_h = (_Float16*)alloc(sizeof(_Float16) * Lc * 128 * 128);
  _Float16* mp_wp2_h = (_Float16*)alloc(sizeof(_Float16) * Lc * 128 * 384);
  _Float16* mp_wf_h  = (_Float16*)alloc(sizeof(_Float16) * Lc * 128 * 384);
  _Float16* up_u_h   = (_Float16*)alloc(sizeof(_Float16) * Lc * 128 * 128);
  _Float16* up_v_h   = (_Float16*)alloc(sizeof(_Float16) * Lc * 128 * 128);
  _Float16* up_w1_h  = (_Float16*)alloc(sizeof(_Float16) * Lc * 256 * 128);
  _Float16* up_w2_h  = (_Float16*)alloc(sizeof(_Float16) * Lc * 128 * 384);
  _Float16* lg_ws_h  = (_Float16*)alloc(sizeof(_Float16) * Lc * 128 * 128);
  _Float16* lg_wv_h  = (_Float16*)alloc(sizeof(_Float16) * Lc * 128 * 128);

  auto cdiv = [](long long a, long long b) { return (unsigned)((a + b - 1) / b); };
#define GS(n) dim3(cdiv((n), 256)), dim3(256), 0, stream

  auto cvt = [&](const float* s, _Float16* d, long long n) {
    k_scale_f16<<<GS(n)>>>(s, d, 1.0f, n);
  };
  auto gemm = [&](const _Float16* A, const _Float16* B, const float* bias,
                  float* Cf, _Float16* Ch, int M, int Ncol, int K, bool silu) {
    dim3 grid((unsigned)((M + 31) / 32), (unsigned)(Ncol / 128));
    if (silu) gemm_wmma<true><<<grid, 256, 0, stream>>>(A, B, bias, Cf, Ch, M, Ncol, K);
    else      gemm_wmma<false><<<grid, 256, 0, stream>>>(A, B, bias, Cf, Ch, M, Ncol, K);
  };

  // --- one-time weight conversion (deterministic each call) ---
  cvt(fm_w1, fm_w1_h, (long long)Lc * 256 * 128);
  cvt(fm_w2, fm_w2_h, (long long)Lc * 128 * 128);
  cvt(lb_wf, lb_wf_h, (long long)Lc * 128 * 384);
  cvt(lb_wp1, lb_wp1_h, (long long)Lc * 128 * 128);
  cvt(lb_wp2, lb_wp2_h, (long long)Lc * 128 * 384);
  cvt(mp_wp1, mp_wp1_h, (long long)Lc * 128 * 128);
  cvt(mp_wp2, mp_wp2_h, (long long)Lc * 128 * 384);
  cvt(mp_wf, mp_wf_h, (long long)Lc * 128 * 384);
  cvt(up_u, up_u_h, (long long)Lc * 128 * 128);
  cvt(up_v, up_v_h, (long long)Lc * 128 * 128);
  cvt(up_w1, up_w1_h, (long long)Lc * 256 * 128);
  cvt(up_w2, up_w2_h, (long long)Lc * 128 * 384);
  cvt(lg_ws, lg_ws_h, (long long)Lc * 128 * 128);
  cvt(lg_wv, lg_wv_h, (long long)Lc * 128 * 128);

  // --- state init ---
  k_copy<<<GS((long long)Nc * 3)>>>(pos_cur, pos0, (long long)Nc * 3);
  k_copy<<<GS((long long)Gc * 9)>>>(cell_cur, cell0, (long long)Gc * 9);
  k_init_atom_s<<<GS((long long)Nc * Hc)>>>(z, atom_emb, atom_s, (long long)Nc * Hc);
  k_zero<<<GS((long long)Nc * 3 * Hc)>>>(atom_v, (long long)Nc * 3 * Hc);
  k_init_lat_s<<<GS((long long)Gc * 3 * Hc)>>>(lat_emb, lat_s, (long long)Gc * 3 * Hc);
  k_zero<<<GS((long long)Gc * 9 * Hc)>>>(lat_v, (long long)Gc * 9 * Hc);

  for (int n = 0; n < Lc; ++n) {
    // --- edge geometry + RBF ---
    k_edge_geom<<<dim3(Ec), dim3(128), 0, stream>>>(pos_cur, cell_cur, offs, eidx, batch,
                                                    udiff, efeat_h);
    // --- per-graph pos mean (as sum; divided at use site) ---
    k_zero<<<GS((long long)Gc * 3)>>>(pos_sum, (long long)Gc * 3);
    k_pos_accum<<<GS((long long)Nc)>>>(pos_cur, batch, pos_sum, (long long)Nc);
    // --- lattice per-atom features ---
    k_lat_atom<<<dim3(Nc * 3), dim3(128), 0, stream>>>(pos_cur, pos_sum, cell_cur, batch,
                                                       lat_ud, latrbf_h);
    // --- lattice feature MLP + filter GEMMs ---
    gemm(latrbf_h, fm_w1_h + (size_t)n * 256 * 128, fm_b1 + n * 128,
         nullptr, lat_h1_h, Nc * 3, 128, 256, true);
    gemm(lat_h1_h, fm_w2_h + (size_t)n * 128 * 128, fm_b2 + n * 128,
         nullptr, lat_ft_h, Nc * 3, 128, 128, false);
    gemm(lat_ft_h, lb_wf_h + (size_t)n * 128 * 384, lb_bf + n * 384,
         filt, nullptr, Nc * 3, 384, 128, false);
    // --- per-graph lattice phi MLP ---
    cvt(lat_s, lats_h, (long long)Gc * 3 * Hc);
    gemm(lats_h, lb_wp1_h + (size_t)n * 128 * 128, lb_bp1 + n * 128,
         nullptr, lbp_t_h, Gc * 3, 128, 128, true);
    gemm(lbp_t_h, lb_wp2_h + (size_t)n * 128 * 384, lb_bp2 + n * 384,
         phi_lb, nullptr, Gc * 3, 384, 128, false);
    // --- lattice -> atoms ---
    k_lb_apply<<<dim3(Nc), dim3(128), 0, stream>>>(filt, phi_lb, lat_v, lat_ud, batch,
                                                   atom_s, atom_v);
    // --- message passing ---
    cvt(atom_s, atoms_h, (long long)Nc * Hc);
    gemm(atoms_h, mp_wp1_h + (size_t)n * 128 * 128, mp_bp1 + n * 128,
         nullptr, mp_t_h, Nc, 128, 128, true);
    gemm(mp_t_h, mp_wp2_h + (size_t)n * 128 * 384, mp_bp2 + n * 384,
         phi_mp, nullptr, Nc, 384, 128, false);
    gemm(efeat_h, mp_wf_h + (size_t)n * 128 * 384, mp_bf + n * 384,
         nullptr, Wf_h, Ec, 384, 128, false);
    k_zero<<<GS((long long)Nc * Hc)>>>(d_s, (long long)Nc * Hc);
    k_zero<<<GS((long long)Nc * 3 * Hc)>>>(d_v, (long long)Nc * 3 * Hc);
    k_message<<<dim3(Ec), dim3(128), 0, stream>>>(phi_mp, Wf_h, atom_v, udiff, eidx, d_s, d_v);
    k_mp_apply_s<<<GS((long long)Nc * Hc)>>>(atom_s, d_s, (long long)Nc * Hc);
    k_add<<<GS((long long)Nc * 3 * Hc)>>>(atom_v, d_v, (long long)Nc * 3 * Hc);
    // --- self-interaction update ---
    cvt(atom_v, atomv_h, (long long)Nc * 3 * Hc);
    gemm(atomv_h, up_u_h + (size_t)n * 128 * 128, nullptr, Uv, nullptr, Nc * 3, 128, 128, false);
    gemm(atomv_h, up_v_h + (size_t)n * 128 * 128, nullptr, Vv, nullptr, Nc * 3, 128, 128, false);
    k_vnorm_cat<<<GS((long long)Nc * Hc)>>>(atom_s, Vv, cat_h, (long long)Nc * Hc);
    gemm(cat_h, up_w1_h + (size_t)n * 256 * 128, up_b1 + n * 128,
         nullptr, up_t_h, Nc, 128, 256, true);
    gemm(up_t_h, up_w2_h + (size_t)n * 128 * 384, up_b2 + n * 384,
         m_up, nullptr, Nc, 384, 128, false);
    k_update_apply<<<GS((long long)Nc * Hc)>>>(atom_s, atom_v, Uv, Vv, m_up, (long long)Nc * Hc);
    // --- structure update ---
    k_structure<<<GS((long long)Nc * 3)>>>(pos_cur, atom_v, su_w + n * 128, (long long)Nc * 3);
    // --- graph pooling + lattice decode ---
    k_zero<<<GS((long long)Gc * Hc)>>>(s_pool, (long long)Gc * Hc);
    k_zero<<<GS((long long)Gc * 3 * Hc)>>>(v_pool, (long long)Gc * 3 * Hc);
    k_pool<<<GS((long long)Nc * Hc)>>>(atom_s, atom_v, batch, s_pool, v_pool, (long long)Nc * Hc);
    k_scale_f16<<<GS((long long)Gc * Hc)>>>(s_pool, spool_h, 1.0f / APGc, (long long)Gc * Hc);
    k_scale_f16<<<GS((long long)Gc * 3 * Hc)>>>(v_pool, vpool_h, 1.0f / APGc, (long long)Gc * 3 * Hc);
    gemm(spool_h, lg_ws_h + (size_t)n * 128 * 128, lg_bs + n * 128,
         t_ls, nullptr, Gc, 128, 128, true);
    gemm(vpool_h, lg_wv_h + (size_t)n * 128 * 128, nullptr,
         vproj, nullptr, Gc * 3, 128, 128, false);
    k_lat_apply<<<dim3(Gc), dim3(128), 0, stream>>>(lat_s, lat_v, cell_cur, t_ls, vproj,
                                                    ldec_w + n * 128);
    // --- append outputs: pos stack then cell stack ---
    k_copy<<<GS((long long)Nc * 3)>>>(out + (size_t)n * Nc * 3, pos_cur, (long long)Nc * 3);
    k_copy<<<GS((long long)Gc * 9)>>>(out + (size_t)Lc * Nc * 3 + (size_t)n * Gc * 9,
                                      cell_cur, (long long)Gc * 9);
  }
#undef GS
}
